// SS3DScan_69423851372928
// MI455X (gfx1250) — compile-verified, matrix operand order checked by reading, not run on hardware
//
#include <hip/hip_runtime.h>
#include <hip/hip_bf16.h>

typedef __attribute__((ext_vector_type(16))) _Float16 v16h;
typedef __attribute__((ext_vector_type(8)))  _Float16 v8h;
typedef __attribute__((ext_vector_type(4)))  _Float16 v4h;
typedef __attribute__((ext_vector_type(8)))  float    v8f;
typedef __attribute__((ext_vector_type(4)))  float    v4f;

#define L_TOT  13824      // 24*24*24
#define NCH    216        // chunks of 64
#define DM_    256
#define DI_    512
#define NH_    8
#define NPAD   1280       // 1160 padded up to 10*128
#define NPCOLS 1160       // 2*DI + 2*N + NH

// ---------------------------------------------------------------------------
// WMMA helpers (CDNA5 16x16x32 f16 -> f32)
// ---------------------------------------------------------------------------
__device__ __forceinline__ v8f wmma32(v16h a, v16h b, v8f c) {
  return __builtin_amdgcn_wmma_f32_16x16x32_f16(false, a, false, b, (short)0, c,
                                                false, false);
}

// Fragment loader: two contiguous b128 LDS reads per 16-bit operand fragment.
template <int LD>
__device__ __forceinline__ v16h frag_row(const _Float16 S[][LD], int tile, int kk, int lane) {
  int r = tile * 16 + (lane & 15), hi = lane >> 4;
  v8h lo = *reinterpret_cast<const v8h*>(&S[r][kk + 8 * hi]);
  v8h hiv = *reinterpret_cast<const v8h*>(&S[r][kk + 16 + 8 * hi]);
  return __builtin_shufflevector(lo, hiv, 0, 1, 2, 3, 4, 5, 6, 7,
                                 8, 9, 10, 11, 12, 13, 14, 15);
}

// ---------------------------------------------------------------------------
// CDNA5 async global->LDS b128 copy (ASYNCcnt-tracked, no VGPR round trip).
// Flat LDS address truncated to 32 bits is the architected LDS byte offset.
// ---------------------------------------------------------------------------
__device__ __forceinline__ void async_ld_b128(void* lds_ptr, const void* gptr) {
  unsigned lds = (unsigned)(size_t)lds_ptr;
  asm volatile("global_load_async_to_lds_b128 %0, %1, off"
               :: "v"(lds), "v"(gptr) : "memory");
}
__device__ __forceinline__ void wait_async_le3() {
  asm volatile("s_wait_asynccnt 0x3" ::: "memory");
}
__device__ __forceinline__ void wait_async_0() {
  asm volatile("s_wait_asynccnt 0x0" ::: "memory");
}

// ---------------------------------------------------------------------------
// direction mapping: voxel index for sequence position t
// ---------------------------------------------------------------------------
__device__ __forceinline__ int voxel_of_t(int t, int dir) {
  int tt = (dir & 1) ? (L_TOT - 1 - t) : t;
  int a = tt / 576, b = (tt / 24) % 24, c = tt % 24;
  int d, h, w;
  switch (dir >> 1) {
    case 0:  d = a; h = b; w = c; break;   // (D,H,W)
    case 1:  h = a; d = b; w = c; break;   // (H,D,W)
    default: w = a; d = b; h = c; break;   // (W,D,H)
  }
  return (d * 24 + h) * 24 + w;
}

// ---------------------------------------------------------------------------
// weight f32 -> f16, TRANSPOSED with column padding: dstT[colPad][rows]
// ---------------------------------------------------------------------------
__global__ void k_cvt_padT(const float* __restrict__ src, _Float16* __restrict__ dstT,
                           int rows, int cols, int colsPad) {
  int i = blockIdx.x * blockDim.x + threadIdx.x;
  if (i >= rows * colsPad) return;
  int c = i / rows, r = i % rows;
  dstT[(size_t)c * rows + r] = (c < cols) ? (_Float16)src[(size_t)r * cols + c]
                                          : (_Float16)0.f;
}

// gather permuted sequence into f16 GEMM operand, 8 channels/thread
__global__ void k_gather(const float* __restrict__ x, _Float16* __restrict__ xseq, int dir) {
  int i = blockIdx.x * blockDim.x + threadIdx.x;
  if (i >= L_TOT * (DM_ / 8)) return;
  int t = i >> 5, cg = (i & 31) * 8;
  const float* src = &x[(size_t)voxel_of_t(t, dir) * DM_ + cg];
  v4f a = *reinterpret_cast<const v4f*>(src);
  v4f b = *reinterpret_cast<const v4f*>(src + 4);
  v8h o;
#pragma unroll
  for (int e = 0; e < 4; ++e) { o[e] = (_Float16)a[e]; o[e + 4] = (_Float16)b[e]; }
  *reinterpret_cast<v8h*>(&xseq[(size_t)t * DM_ + cg]) = o;
}

// ---------------------------------------------------------------------------
// WMMA GEMM: C[M,N] = A[M,K](f16,row-major) * BT[N,K](f16, pre-transposed)
// 256 thr = 8 waves; block tile 64x128; wave = 1 A-frag x 4 B-frags = 4 WMMA.
// Async global->LDS double-buffered staging (3 async b128 per thread/stage).
// M%64==0, N%128==0, K%32==0.
// ---------------------------------------------------------------------------
__global__ void __launch_bounds__(256) k_gemm_wmma(
    const _Float16* __restrict__ A, const _Float16* __restrict__ BT,
    float* __restrict__ Cmat, int M, int N, int K) {
  __shared__ __align__(16) _Float16 As[2][64][40];
  __shared__ __align__(16) _Float16 Bs[2][128][40];
  int tid = threadIdx.x, lane = tid & 31, wid = tid >> 5;
  int m0 = blockIdx.y * 64, n0 = blockIdx.x * 128;
  int mt = wid & 3, ntB = (wid >> 2) * 4;
  int ar = tid >> 2, ac = (tid & 3) * 8;        // A stage: 1 v8h / thread
  int br0 = tid >> 2, br1 = (tid + 256) >> 2;   // B stage: 2 v8h / thread
  int bc = (tid & 3) * 8;
  v8f acc[4] = {};

  auto issue = [&](int k0, int buf) {
    async_ld_b128(&As[buf][ar][ac], &A[(size_t)(m0 + ar) * K + k0 + ac]);
    async_ld_b128(&Bs[buf][br0][bc], &BT[(size_t)(n0 + br0) * K + k0 + bc]);
    async_ld_b128(&Bs[buf][br1][bc], &BT[(size_t)(n0 + br1) * K + k0 + bc]);
  };

  issue(0, 0);
  for (int k0 = 0; k0 < K; k0 += 32) {
    int buf = (k0 >> 5) & 1;
    bool hasNext = (k0 + 32) < K;
    if (hasNext) { issue(k0 + 32, buf ^ 1); wait_async_le3(); }
    else         { wait_async_0(); }
    __syncthreads();
    v16h af = frag_row<40>(As[buf], mt, 0, lane);
#pragma unroll
    for (int q = 0; q < 4; ++q) {
      v16h bq = frag_row<40>(Bs[buf], ntB + q, 0, lane);
      acc[q] = wmma32(af, bq, acc[q]);
    }
    __syncthreads();
  }
#pragma unroll
  for (int q = 0; q < 4; ++q) {
    int pcol = (ntB + q) * 16 + (lane & 15);
#pragma unroll
    for (int r = 0; r < 8; ++r) {
      int mrow = m0 + mt * 16 + r + 8 * (lane >> 4);
      Cmat[(size_t)mrow * N + n0 + pcol] = acc[q][r];
    }
  }
}

// ---------------------------------------------------------------------------
// depthwise causal conv (window 4) + SiLU, 4 channels/thread
// ---------------------------------------------------------------------------
__global__ void k_conv_silu(const float* __restrict__ zx, const float* __restrict__ cw,
                            const float* __restrict__ cb, float* __restrict__ xact) {
  int i = blockIdx.x * blockDim.x + threadIdx.x;
  if (i >= L_TOT * 160) return;
  int t = i / 160, cg = (i % 160) * 4;
  v4f w0 = *reinterpret_cast<const v4f*>(&cw[(cg + 0) * 4]);
  v4f w1 = *reinterpret_cast<const v4f*>(&cw[(cg + 1) * 4]);
  v4f w2 = *reinterpret_cast<const v4f*>(&cw[(cg + 2) * 4]);
  v4f w3 = *reinterpret_cast<const v4f*>(&cw[(cg + 3) * 4]);
  v4f s = *reinterpret_cast<const v4f*>(&cb[cg]);
#pragma unroll
  for (int k = 0; k < 4; ++k) {
    int tt = t - 3 + k;
    if (tt >= 0) {
      v4f xv = *reinterpret_cast<const v4f*>(&zx[(size_t)tt * NPAD + 512 + cg]);
      s[0] += xv[0] * w0[k];
      s[1] += xv[1] * w1[k];
      s[2] += xv[2] * w2[k];
      s[3] += xv[3] * w3[k];
    }
  }
  v4f o;
#pragma unroll
  for (int e = 0; e < 4; ++e) o[e] = s[e] / (1.f + __expf(-s[e]));
  *reinterpret_cast<v4f*>(&xact[(size_t)t * 640 + cg]) = o;
}

// softplus dt + per-chunk inclusive scan of dt*A (log decays)
__global__ void k_chunkscan(const float* __restrict__ zx, const float* __restrict__ dt_bias,
                            const float* __restrict__ A_log, float* __restrict__ dtarr,
                            float* __restrict__ Acum, float* __restrict__ Asum) {
  int c = blockIdx.x, h = blockIdx.y, tl = threadIdx.x;  // blockDim = 64
  int t = c * 64 + tl;
  float raw = zx[(size_t)t * NPAD + 1152 + h] + dt_bias[h];
  float dt = (raw > 20.f) ? raw : log1pf(__expf(raw));
  float a = -__expf(A_log[h]) * dt;
  __shared__ float s[64];
  s[tl] = a;
  __syncthreads();
  for (int off = 1; off < 64; off <<= 1) {
    float v = (tl >= off) ? s[tl - off] : 0.f;
    __syncthreads();
    s[tl] += v;
    __syncthreads();
  }
  dtarr[t * 8 + h] = dt;
  Acum[t * 8 + h] = s[tl];
  if (tl == 63) Asum[c * 8 + h] = s[63];
}

// XsT[(chunk,head)][p][t] = x*dt (f16, transposed for SSD), B/C f16
__global__ void k_prep16(const float* __restrict__ xact, const float* __restrict__ dtarr,
                         _Float16* __restrict__ XsT, _Float16* __restrict__ B16,
                         _Float16* __restrict__ C16) {
  int i = blockIdx.x * blockDim.x + threadIdx.x;
  if (i >= L_TOT * 160) return;
  int t = i / 160, cg = (i % 160) * 4;
  v4f v = *reinterpret_cast<const v4f*>(&xact[(size_t)t * 640 + cg]);
  if (cg < 512) {
    int h = cg >> 6, c = t >> 6, tl = t & 63;
    float dt = dtarr[t * 8 + h];
#pragma unroll
    for (int e = 0; e < 4; ++e) {
      int p = (cg & 63) + e;
      XsT[((size_t)(c * 8 + h) * 64 + p) * 64 + tl] = (_Float16)(v[e] * dt);
    }
  } else {
    v4h o;
#pragma unroll
    for (int e = 0; e < 4; ++e) o[e] = (_Float16)v[e];
    if (cg < 576)
      *reinterpret_cast<v4h*>(&B16[(size_t)t * 64 + (cg - 512)]) = o;
    else
      *reinterpret_cast<v4h*>(&C16[(size_t)t * 64 + (cg - 576)]) = o;
  }
}

// ---------------------------------------------------------------------------
// SSD intra-chunk. One block per (chunk, head), 8 waves. Async LDS staging.
// ---------------------------------------------------------------------------
__global__ void __launch_bounds__(256) k_ssd_intra(
    const _Float16* __restrict__ B16, const _Float16* __restrict__ C16,
    const _Float16* __restrict__ XsT, const float* __restrict__ Acum,
    float* __restrict__ y_pre, float* __restrict__ states) {
  int c = blockIdx.x, h = blockIdx.y;
  int tid = threadIdx.x, lane = tid & 31, wid = tid >> 5;
  __shared__ __align__(16) _Float16 Cs[64][72];   // [t][n]
  __shared__ __align__(16) _Float16 Bs[64][72];   // [t][n]
  __shared__ __align__(16) _Float16 Xsh[64][72];  // [p][t]
  __shared__ __align__(16) _Float16 Gs[64][72];   // phase1: [t][s]; phase3: BdT [n][t]
  __shared__ float Ac[64];
#pragma unroll
  for (int j = 0; j < 2; ++j) {
    int idx = tid + j * 256;
    int r = idx >> 3, cg = (idx & 7) * 8;
    int tg = c * 64 + r;
    async_ld_b128(&Cs[r][cg], &C16[(size_t)tg * 64 + cg]);
    async_ld_b128(&Bs[r][cg], &B16[(size_t)tg * 64 + cg]);
    async_ld_b128(&Xsh[r][cg], &XsT[((size_t)(c * 8 + h) * 64 + r) * 64 + cg]);
  }
  if (tid < 64) Ac[tid] = Acum[(size_t)(c * 64 + tid) * 8 + h];
  wait_async_0();
  __syncthreads();
  int mt = wid & 3, ntB = (wid >> 2) * 2;

  // phase 1: G = C * B^T (K over n), mask+exp -> Gs[t][s] (f16)
#pragma unroll
  for (int p = 0; p < 2; ++p) {
    v8f a = {};
    for (int kk = 0; kk < 64; kk += 32) {
      v16h af = frag_row<72>(Cs, mt, kk, lane);
      v16h bf = frag_row<72>(Bs, ntB + p, kk, lane);  // B_op[n][s] = Bs[s][n]
      a = wmma32(af, bf, a);
    }
    int scol = (ntB + p) * 16 + (lane & 15);
#pragma unroll
    for (int r = 0; r < 8; ++r) {
      int trow = mt * 16 + r + 8 * (lane >> 4);
      float g = (scol <= trow) ? __expf(Ac[trow] - Ac[scol]) * a[r] : 0.f;
      Gs[trow][scol] = (_Float16)g;
    }
  }
  __syncthreads();

  // phase 2: Y_diag = Gs * X (K over s);  B_op[s][p] = Xsh[p][s]
#pragma unroll
  for (int p = 0; p < 2; ++p) {
    v8f a = {};
    for (int kk = 0; kk < 64; kk += 32) {
      v16h af = frag_row<72>(Gs, mt, kk, lane);
      v16h bf = frag_row<72>(Xsh, ntB + p, kk, lane);
      a = wmma32(af, bf, a);
    }
    int pcol = (ntB + p) * 16 + (lane & 15);
#pragma unroll
    for (int r = 0; r < 8; ++r) {
      int trow = mt * 16 + r + 8 * (lane >> 4);
      y_pre[((size_t)(c * 64 + trow) * 8 + h) * 64 + pcol] = a[r];
    }
  }
  __syncthreads();

  // phase 3: BdT[n][t] = B[t][n]*exp(Ac63-Ac[t]);  statesT[p][n] = X^T * Bd
  for (int i = tid; i < 4096; i += 256) {
    int tl = i >> 6, n = i & 63;
    Gs[n][tl] = (_Float16)((float)Bs[tl][n] * __expf(Ac[63] - Ac[tl]));
  }
  __syncthreads();
#pragma unroll
  for (int p = 0; p < 2; ++p) {
    v8f a = {};
    for (int kk = 0; kk < 64; kk += 32) {
      v16h af = frag_row<72>(Xsh, mt, kk, lane);      // A_op[p][t] = Xsh[p][t]
      v16h bf = frag_row<72>(Gs, ntB + p, kk, lane);  // B_op[t][n] = Gs[n][t]
      a = wmma32(af, bf, a);
    }
    int ncol = (ntB + p) * 16 + (lane & 15);
#pragma unroll
    for (int r = 0; r < 8; ++r) {
      int prow = mt * 16 + r + 8 * (lane >> 4);
      states[((size_t)(c * 8 + h) * 64 + prow) * 64 + ncol] = a[r];  // [p][n]
    }
  }
}

// inter-chunk recurrence on statesT[p][n]: elementwise, float4 rows
__global__ void k_ssd_inter(const float* __restrict__ states, const float* __restrict__ Asum,
                            float* __restrict__ states_in) {
  int h = blockIdx.x, tid = threadIdx.x;
  v4f S[4];
#pragma unroll
  for (int i = 0; i < 4; ++i) S[i] = (v4f){0.f, 0.f, 0.f, 0.f};
  for (int c = 0; c < NCH; ++c) {
    size_t base = ((size_t)c * 8 + h) * 4096 + (size_t)tid * 16;
    float ef = __expf(Asum[c * 8 + h]);
#pragma unroll
    for (int i = 0; i < 4; ++i) {
      v4f st = *reinterpret_cast<const v4f*>(&states[base + i * 4]);
      *reinterpret_cast<v4f*>(&states_in[base + i * 4]) = S[i];
      S[i] = S[i] * ef + st;
    }
  }
}

// Y_off = exp(Acum)*(C * state_in) + X*D, accumulated into y_pre
__global__ void __launch_bounds__(256) k_ssd_off(
    const _Float16* __restrict__ C16, const float* __restrict__ states_in,
    const float* __restrict__ Acum, const float* __restrict__ xact,
    const float* __restrict__ Dp, float* __restrict__ y_pre) {
  int c = blockIdx.x, h = blockIdx.y;
  int tid = threadIdx.x, lane = tid & 31, wid = tid >> 5;
  __shared__ __align__(16) _Float16 Cs[64][72];   // [t][n]
  __shared__ __align__(16) _Float16 Ssh[64][72];  // [p][n]
  __shared__ float Ac[64];
#pragma unroll
  for (int j = 0; j < 2; ++j) {
    int idx = tid + j * 256;
    int r = idx >> 3, cg = (idx & 7) * 8;
    async_ld_b128(&Cs[r][cg], &C16[(size_t)(c * 64 + r) * 64 + cg]);
    size_t sb = ((size_t)(c * 8 + h) * 64 + r) * 64 + cg;
    v4f s0 = *reinterpret_cast<const v4f*>(&states_in[sb]);
    v4f s1 = *reinterpret_cast<const v4f*>(&states_in[sb + 4]);
    v8h o;
#pragma unroll
    for (int e = 0; e < 4; ++e) { o[e] = (_Float16)s0[e]; o[e + 4] = (_Float16)s1[e]; }
    *reinterpret_cast<v8h*>(&Ssh[r][cg]) = o;
  }
  if (tid < 64) Ac[tid] = Acum[(size_t)(c * 64 + tid) * 8 + h];
  wait_async_0();
  __syncthreads();
  float Dh = Dp[h];
  int mt = wid & 3, ntB = (wid >> 2) * 2;
#pragma unroll
  for (int p = 0; p < 2; ++p) {
    v8f a = {};
    for (int kk = 0; kk < 64; kk += 32) {
      v16h af = frag_row<72>(Cs, mt, kk, lane);
      v16h bf = frag_row<72>(Ssh, ntB + p, kk, lane);  // B_op[n][p] = Ssh[p][n]
      a = wmma32(af, bf, a);
    }
    int pcol = (ntB + p) * 16 + (lane & 15);
#pragma unroll
    for (int r = 0; r < 8; ++r) {
      int trow = mt * 16 + r + 8 * (lane >> 4);
      int tg = c * 64 + trow;
      size_t yi = ((size_t)tg * 8 + h) * 64 + pcol;
      float xr = xact[(size_t)tg * 640 + h * 64 + pcol];
      y_pre[yi] += __expf(Ac[trow]) * a[r] + xr * Dh;
    }
  }
}

// y = Y*silu(z), RMSNorm over DI, *norm_w -> f16 (128 thr, 4 ch each)
__global__ void k_gate_norm(const float* __restrict__ y_pre, const float* __restrict__ zx,
                            const float* __restrict__ norm_w, _Float16* __restrict__ y16) {
  int t = blockIdx.x, tid = threadIdx.x;  // blockDim = 128
  __shared__ float red[128];
  int cg = tid * 4;
  v4f y = *reinterpret_cast<const v4f*>(&y_pre[(size_t)t * 512 + cg]);
  v4f z = *reinterpret_cast<const v4f*>(&zx[(size_t)t * NPAD + cg]);
  v4f v;
  float ss = 0.f;
#pragma unroll
  for (int e = 0; e < 4; ++e) {
    v[e] = y[e] * (z[e] / (1.f + __expf(-z[e])));
    ss += v[e] * v[e];
  }
  red[tid] = ss;
  __syncthreads();
  for (int o = 64; o > 0; o >>= 1) {
    if (tid < o) red[tid] += red[tid + o];
    __syncthreads();
  }
  float scale = rsqrtf(red[0] / 512.f + 1e-6f);
  v4f nwv = *reinterpret_cast<const v4f*>(&norm_w[cg]);
  v4h o;
#pragma unroll
  for (int e = 0; e < 4; ++e) o[e] = (_Float16)(v[e] * scale * nwv[e]);
  *reinterpret_cast<v4h*>(&y16[(size_t)t * 512 + cg]) = o;
}

// softmax gates per voxel
__global__ void k_gates(const float* __restrict__ x, const float* __restrict__ gw,
                        const float* __restrict__ gb, float* __restrict__ gates) {
  int v = blockIdx.x * blockDim.x + threadIdx.x;
  if (v >= L_TOT) return;
  float lg[6];
#pragma unroll
  for (int d = 0; d < 6; ++d) lg[d] = gb[d];
  for (int c = 0; c < 256; ++c) {
    float xv = x[(size_t)v * 256 + c];
#pragma unroll
    for (int d = 0; d < 6; ++d) lg[d] += xv * gw[c * 6 + d];
  }
  float m = lg[0];
#pragma unroll
  for (int d = 1; d < 6; ++d) m = fmaxf(m, lg[d]);
  float s = 0.f;
#pragma unroll
  for (int d = 0; d < 6; ++d) { lg[d] = __expf(lg[d] - m); s += lg[d]; }
#pragma unroll
  for (int d = 0; d < 6; ++d) gates[(size_t)v * 6 + d] = lg[d] / s;
}

// gated scatter into output, 4 channels/thread (dir 0 initializes)
__global__ void k_scatter(const float* __restrict__ out_dir, const float* __restrict__ gates,
                          float* __restrict__ out, int dir) {
  int i = blockIdx.x * blockDim.x + threadIdx.x;
  if (i >= L_TOT * 64) return;
  int t = i >> 6, cg = (i & 63) * 4;
  int v = voxel_of_t(t, dir);
  float g = gates[(size_t)v * 6 + dir];
  v4f val = *reinterpret_cast<const v4f*>(&out_dir[(size_t)t * 256 + cg]);
  float* op = &out[(size_t)v * 256 + cg];
  if (dir == 0) {
#pragma unroll
    for (int e = 0; e < 4; ++e) val[e] *= g;
    *reinterpret_cast<v4f*>(op) = val;
  } else {
    v4f cur = *reinterpret_cast<const v4f*>(op);
#pragma unroll
    for (int e = 0; e < 4; ++e) cur[e] += g * val[e];
    *reinterpret_cast<v4f*>(op) = cur;
  }
}

// ---------------------------------------------------------------------------
extern "C" void kernel_launch(void* const* d_in, const int* in_sizes, int n_in,
                              void* d_out, int out_size, void* d_ws, size_t ws_size,
                              hipStream_t stream) {
  (void)in_sizes; (void)n_in; (void)out_size; (void)ws_size;
  const float* x = (const float*)d_in[0];
  const float* inproj[3]  = {(const float*)d_in[1],  (const float*)d_in[9],  (const float*)d_in[17]};
  const float* convw[3]   = {(const float*)d_in[2],  (const float*)d_in[10], (const float*)d_in[18]};
  const float* convb[3]   = {(const float*)d_in[3],  (const float*)d_in[11], (const float*)d_in[19]};
  const float* dtb[3]     = {(const float*)d_in[4],  (const float*)d_in[12], (const float*)d_in[20]};
  const float* alog[3]    = {(const float*)d_in[5],  (const float*)d_in[13], (const float*)d_in[21]};
  const float* Dp[3]      = {(const float*)d_in[6],  (const float*)d_in[14], (const float*)d_in[22]};
  const float* nw[3]      = {(const float*)d_in[7],  (const float*)d_in[15], (const float*)d_in[23]};
  const float* outproj[3] = {(const float*)d_in[8],  (const float*)d_in[16], (const float*)d_in[24]};
  const float* gw = (const float*)d_in[25];
  const float* gb = (const float*)d_in[26];

  size_t off = 0;
  char* base = (char*)d_ws;
  auto carve = [&](size_t bytes) -> void* {
    void* p = base + off;
    off += (bytes + 255) & ~(size_t)255;
    return p;
  };
  _Float16* WinT  = (_Float16*)carve((size_t)3 * NPAD * 256 * 2);  // [n][k]
  _Float16* WoutT = (_Float16*)carve((size_t)3 * 256 * 512 * 2);   // [n][k]
  float* gates     = (float*)carve((size_t)L_TOT * 6 * 4);
  _Float16* xseq16 = (_Float16*)carve((size_t)L_TOT * 256 * 2);
  float* zx        = (float*)carve((size_t)L_TOT * NPAD * 4);
  float* xact      = (float*)carve((size_t)L_TOT * 640 * 4);
  float* dtarr     = (float*)carve((size_t)L_TOT * 8 * 4);
  float* Acum      = (float*)carve((size_t)L_TOT * 8 * 4);
  float* Asum      = (float*)carve((size_t)NCH * 8 * 4);
  _Float16* XsT    = (_Float16*)carve((size_t)L_TOT * 512 * 2);    // [(c,h)][p][t]
  _Float16* B16    = (_Float16*)carve((size_t)L_TOT * 64 * 2);
  _Float16* C16    = (_Float16*)carve((size_t)L_TOT * 64 * 2);
  float* states    = (float*)carve((size_t)NCH * 8 * 4096 * 4);    // [p][n]
  float* states_in = (float*)carve((size_t)NCH * 8 * 4096 * 4);
  float* y_pre     = (float*)carve((size_t)L_TOT * 512 * 4);
  _Float16* y16    = (_Float16*)carve((size_t)L_TOT * 512 * 2);
  float* out_dir   = (float*)carve((size_t)L_TOT * 256 * 4);

  // one-time: transposed f16 weights + gates
  for (int ps = 0; ps < 3; ++ps) {
    int tot_in = 256 * NPAD;
    k_cvt_padT<<<(tot_in + 255) / 256, 256, 0, stream>>>(inproj[ps],
        WinT + (size_t)ps * NPAD * 256, 256, NPCOLS, NPAD);
    int tot_out = 512 * 256;
    k_cvt_padT<<<(tot_out + 255) / 256, 256, 0, stream>>>(outproj[ps],
        WoutT + (size_t)ps * 256 * 512, 512, 256, 256);
  }
  k_gates<<<(L_TOT + 255) / 256, 256, 0, stream>>>(x, gw, gb, gates);

  dim3 gCH(NCH, NH_);

  for (int dir = 0; dir < 6; ++dir) {
    int ps = dir >> 1;
    k_gather<<<(L_TOT * 32 + 255) / 256, 256, 0, stream>>>(x, xseq16, dir);

    dim3 g1(NPAD / 128, L_TOT / 64);
    k_gemm_wmma<<<g1, 256, 0, stream>>>(xseq16, WinT + (size_t)ps * NPAD * 256,
                                        zx, L_TOT, NPAD, 256);

    k_conv_silu<<<(L_TOT * 160 + 255) / 256, 256, 0, stream>>>(zx, convw[ps], convb[ps], xact);
    k_chunkscan<<<gCH, 64, 0, stream>>>(zx, dtb[ps], alog[ps], dtarr, Acum, Asum);
    k_prep16<<<(L_TOT * 160 + 255) / 256, 256, 0, stream>>>(xact, dtarr, XsT, B16, C16);

    k_ssd_intra<<<gCH, 256, 0, stream>>>(B16, C16, XsT, Acum, y_pre, states);
    k_ssd_inter<<<NH_, 256, 0, stream>>>(states, Asum, states_in);
    k_ssd_off<<<gCH, 256, 0, stream>>>(C16, states_in, Acum, xact, Dp[ps], y_pre);

    k_gate_norm<<<L_TOT, 128, 0, stream>>>(y_pre, zx, nw[ps], y16);

    dim3 g2(256 / 128, L_TOT / 64);
    k_gemm_wmma<<<g2, 256, 0, stream>>>(y16, WoutT + (size_t)ps * 256 * 512,
                                        out_dir, L_TOT, 256, 512);

    k_scatter<<<(L_TOT * 64 + 255) / 256, 256, 0, stream>>>(out_dir, gates,
                                                            (float*)d_out, dir);
  }
}
